// MixtureOfExperts_83365315215461
// MI455X (gfx1250) — matrix operand order repaired and checked by reading, not compiled, not run on hardware
//
#include <hip/hip_runtime.h>
#include <hip/hip_bf16.h>

// ---------------- problem constants ----------------
#define MOE_T   4096      // B*S tokens
#define MOE_D   1024      // d_model
#define MOE_F   4096      // d_ff
#define MOE_E   8         // routed experts (expert 8 = shared)
#define MOE_NE  9         // experts incl. shared
#define MOE_KC  32        // K-slab depth per WMMA stage
#define MOE_LDA 40        // LDS row stride (bf16 elems), padded vs 32 to break bank conflicts

typedef __bf16 bf16;
typedef __bf16 v8bf  __attribute__((ext_vector_type(8)));
typedef __bf16 v16bf __attribute__((ext_vector_type(16)));
typedef float  v8f   __attribute__((ext_vector_type(8)));

static __device__ inline v16bf moe_cat16(v8bf lo, v8bf hi) {
  return __builtin_shufflevector(lo, hi, 0,1,2,3,4,5,6,7,8,9,10,11,12,13,14,15);
}

static __device__ inline v8f moe_wmma_bf16(v16bf a, v16bf b, v8f c) {
  // D = A(16x32 bf16) * B(32x16 bf16) + C(16x16 f32)
  return __builtin_amdgcn_wmma_f32_16x16x32_bf16(false, a, false, b, (short)0, c,
                                                 false, false);
}

// CDNA5 async global->LDS copy (VGLOBAL encoding, GVS mode: SGPR base + VGPR
// 32-bit byte offset; VDST = LDS byte address). Tracked with ASYNCcnt.
static __device__ inline void moe_async_ld128(unsigned lds_off, unsigned voff,
                                              const void* base) {
  asm volatile("global_load_async_to_lds_b128 %0, %1, %2"
               :
               : "v"(lds_off), "v"(voff), "s"(base)
               : "memory");
}
static __device__ inline void moe_wait_async() {
  asm volatile("s_wait_asynccnt 0x0" ::: "memory");
}
// Low 32 bits of a flat pointer into the LDS aperture == LDS byte offset.
static __device__ inline unsigned moe_lds_off(const void* p) {
  return (unsigned)(unsigned long long)p;
}

// ---------------------------------------------------------------------------
// 1) Router: logits = x @ Wr^T (E=8), softmax -> top2 -> renormalized gates.
//    One wave32 per token; also emits bf16 copy of x for the WMMA GEMMs.
// ---------------------------------------------------------------------------
__global__ __launch_bounds__(256)
void moe_router_kernel(const float* __restrict__ x, const float* __restrict__ Wr,
                       bf16* __restrict__ xb, int* __restrict__ tki,
                       float* __restrict__ tkg) {
  const int lane = threadIdx.x & 31;
  const int wv   = threadIdx.x >> 5;
  const int t    = blockIdx.x * 8 + wv;

  const float* xr = x + (size_t)t * MOE_D;
  float acc[MOE_E];
#pragma unroll
  for (int e = 0; e < MOE_E; ++e) acc[e] = 0.f;

#pragma unroll 4
  for (int j = 0; j < MOE_D / 32; ++j) {
    const int idx = j * 32 + lane;
    const float xv = xr[idx];
    xb[(size_t)t * MOE_D + idx] = (bf16)xv;
#pragma unroll
    for (int e = 0; e < MOE_E; ++e) acc[e] += xv * Wr[e * MOE_D + idx];
  }
#pragma unroll
  for (int off = 16; off; off >>= 1)
#pragma unroll
    for (int e = 0; e < MOE_E; ++e) acc[e] += __shfl_xor(acc[e], off, 32);

  if (lane == 0) {
    float mx = acc[0];
#pragma unroll
    for (int e = 1; e < MOE_E; ++e) mx = fmaxf(mx, acc[e]);
    float p[MOE_E];
#pragma unroll
    for (int e = 0; e < MOE_E; ++e) p[e] = __expf(acc[e] - mx);
    // top-2 (first index wins ties, matching jax.lax.top_k)
    int a1 = 0;
#pragma unroll
    for (int e = 1; e < MOE_E; ++e) if (p[e] > p[a1]) a1 = e;
    int a2 = (a1 == 0) ? 1 : 0;
#pragma unroll
    for (int e = 0; e < MOE_E; ++e) if (e != a1 && p[e] > p[a2]) a2 = e;
    const float gs = p[a1] + p[a2];   // softmax denom cancels under renorm
    tki[t * 2 + 0] = a1; tki[t * 2 + 1] = a2;
    tkg[t * 2 + 0] = p[a1] / gs; tkg[t * 2 + 1] = p[a2] / gs;
  }
}

// ---------------------------------------------------------------------------
// 2) Deterministic compaction: one wave per expert scans tokens in order,
//    ballot+popcount prefix gives stable slots. Expert 8 = shared (all tokens).
// ---------------------------------------------------------------------------
__global__ __launch_bounds__(32)
void moe_scatter_kernel(const int* __restrict__ tki, const float* __restrict__ tkg,
                        int* __restrict__ ids, float* __restrict__ gts,
                        int* __restrict__ slots, int* __restrict__ counts) {
  const int e = blockIdx.x;
  const int lane = threadIdx.x;
  if (e == MOE_E) {  // shared expert: identity list, gate 1, contribution slot 2
    for (int t = lane; t < MOE_T; t += 32) {
      ids  [MOE_E * MOE_T + t] = t;
      gts  [MOE_E * MOE_T + t] = 1.f;
      slots[MOE_E * MOE_T + t] = 2;
    }
    if (lane == 0) counts[MOE_E] = MOE_T;
    return;
  }
  int base = 0;
  for (int c = 0; c < MOE_T; c += 32) {
    const int t = c + lane;
    const bool p0 = (tki[t * 2 + 0] == e);
    const bool p1 = (tki[t * 2 + 1] == e);
    const bool p  = p0 || p1;
    const unsigned mask = __builtin_amdgcn_ballot_w32(p);
    const int pre = __popc(mask & ((1u << lane) - 1u));
    if (p) {
      const int idx = base + pre;
      ids  [e * MOE_T + idx] = t;
      gts  [e * MOE_T + idx] = p0 ? tkg[t * 2 + 0] : tkg[t * 2 + 1];
      slots[e * MOE_T + idx] = p0 ? 0 : 1;
    }
    base += __popc(mask);
  }
  if (lane == 0) counts[e] = base;
}

__global__ void moe_offsets_kernel(const int* __restrict__ counts,
                                   int* __restrict__ offs) {
  if (threadIdx.x == 0 && blockIdx.x == 0) {
    int o = 0;
    for (int e = 0; e < MOE_NE; ++e) { offs[e] = o; o += counts[e]; }
  }
}

// ---------------------------------------------------------------------------
// 3) Gate/Up SwiGLU grouped GEMM: hidden = silu(Xe Wg^T) * (Xe Wu^T), bf16 out.
//    Block tile 128 tokens x 64 ff-cols; 8 waves each own a 32x32 tile for BOTH
//    matrices (shared A staging). A-tiles stream via async global->LDS (bf16
//    copy, no VGPR round-trip); B-tiles go through VALU for f32->bf16 convert.
// ---------------------------------------------------------------------------
__global__ __launch_bounds__(256)
void moe_gateup_kernel(const bf16* __restrict__ xb,
                       const float* __restrict__ Wg, const float* __restrict__ Wu,
                       const float* __restrict__ sWg, const float* __restrict__ sWu,
                       const int* __restrict__ token_ids,
                       const int* __restrict__ counts, const int* __restrict__ offs,
                       bf16* __restrict__ hidden) {
  const int e     = blockIdx.z;
  const int count = counts[e];
  const int m0    = blockIdx.y * 128;
  if (m0 >= count) return;
  const int n0 = blockIdx.x * 64;

  const float* wg = (e < MOE_E) ? (Wg + (size_t)e * MOE_F * MOE_D) : sWg;
  const float* wu = (e < MOE_E) ? (Wu + (size_t)e * MOE_F * MOE_D) : sWu;
  const int*  ids = token_ids + e * MOE_T;

  __shared__ __align__(16) bf16 As [2][128 * MOE_LDA];
  __shared__ __align__(16) bf16 Bgs[2][ 64 * MOE_LDA];
  __shared__ __align__(16) bf16 Bus[2][ 64 * MOE_LDA];

  const int tid = threadIdx.x;

  auto stage = [&](int buf, int kk) {
    {  // A: 128 rows x 32 k (bf16, gathered by token id); 2x16B async per thread
      const int row = tid >> 1, seg = (tid & 1) * 16;
      const int gm = m0 + row;
      const unsigned loff = moe_lds_off(&As[buf][row * MOE_LDA + seg]);
      if (gm < count) {
        const unsigned voff = (unsigned)(((size_t)ids[gm] * MOE_D + kk + seg) *
                                         sizeof(bf16));
        moe_async_ld128(loff, voff, xb);
        moe_async_ld128(loff + 16u, voff + 16u, xb);
      } else {
        const uint4 z = make_uint4(0, 0, 0, 0);
        uint4* q = (uint4*)(&As[buf][row * MOE_LDA + seg]);
        q[0] = z; q[1] = z;
      }
    }
    {  // Bg/Bu: 64 rows x 32 k, f32 -> bf16 on the fly
      const int r = tid >> 2, c8 = (tid & 3) * 8;
      const float* pg = wg + (size_t)(n0 + r) * MOE_D + kk + c8;
      const float* pu = wu + (size_t)(n0 + r) * MOE_D + kk + c8;
      v8bf vg, vu;
#pragma unroll
      for (int i = 0; i < 8; ++i) { vg[i] = (bf16)pg[i]; vu[i] = (bf16)pu[i]; }
      *(v8bf*)(&Bgs[buf][r * MOE_LDA + c8]) = vg;
      *(v8bf*)(&Bus[buf][r * MOE_LDA + c8]) = vu;
    }
  };

  const int lane = tid & 31;
  const int w  = tid >> 5;
  const int wm = (w >> 1) * 32;   // 0,32,64,96 within 128-token tile
  const int wn = (w & 1) * 32;    // 0,32 within 64-col tile

  const v8f vzero = {0.f, 0.f, 0.f, 0.f, 0.f, 0.f, 0.f, 0.f};
  v8f accG[2][2], accU[2][2];
#pragma unroll
  for (int i = 0; i < 2; ++i)
#pragma unroll
    for (int j = 0; j < 2; ++j) { accG[i][j] = vzero; accU[i][j] = vzero; }

  // A frag: lanes 0-15 carry M rows with K 0..7 / 16..23; lanes 16-31 K 8..15 / 24..31
  auto loadA = [&](int buf, int mt) -> v16bf {
    const int row = wm + mt * 16 + (lane & 15);
    const int kb  = (lane >> 4) * 8;
    const bf16* base = &As[buf][row * MOE_LDA + kb];
    return moe_cat16(*(const v8bf*)(base), *(const v8bf*)(base + 16));
  };
  // B frag: lane carries column N=(lane&15); lanes 0-15 K 0..7 / 16..23; lanes 16-31 K 8..15 / 24..31
  auto loadB = [&](const bf16* Bb, int nt) -> v16bf {
    const int rr = wn + nt * 16 + (lane & 15);
    const int kb = (lane >> 4) * 8;
    const bf16* base = Bb + rr * MOE_LDA + kb;
    return moe_cat16(*(const v8bf*)(base), *(const v8bf*)(base + 16));
  };

  stage(0, 0);
  moe_wait_async();
  __syncthreads();
  const int KS = MOE_D / MOE_KC;   // 32 K-steps
  for (int ks = 0; ks < KS; ++ks) {
    const int cur = ks & 1;
    if (ks + 1 < KS) stage(cur ^ 1, (ks + 1) * MOE_KC);
    v16bf a0 = loadA(cur, 0), a1 = loadA(cur, 1);
    v16bf g0 = loadB(&Bgs[cur][0], 0), g1 = loadB(&Bgs[cur][0], 1);
    v16bf u0 = loadB(&Bus[cur][0], 0), u1 = loadB(&Bus[cur][0], 1);
    accG[0][0] = moe_wmma_bf16(a0, g0, accG[0][0]);
    accG[0][1] = moe_wmma_bf16(a0, g1, accG[0][1]);
    accG[1][0] = moe_wmma_bf16(a1, g0, accG[1][0]);
    accG[1][1] = moe_wmma_bf16(a1, g1, accG[1][1]);
    accU[0][0] = moe_wmma_bf16(a0, u0, accU[0][0]);
    accU[0][1] = moe_wmma_bf16(a0, u1, accU[0][1]);
    accU[1][0] = moe_wmma_bf16(a1, u0, accU[1][0]);
    accU[1][1] = moe_wmma_bf16(a1, u1, accU[1][1]);
    moe_wait_async();
    __syncthreads();
  }

  // Epilogue: SwiGLU, cast to bf16, store compacted hidden rows.
  const int hbase = offs[e];
#pragma unroll
  for (int i = 0; i < 2; ++i)
#pragma unroll
    for (int j = 0; j < 2; ++j)
#pragma unroll
      for (int r = 0; r < 8; ++r) {
        const int lrow = wm + i * 16 + r + ((lane >> 4) * 8);
        const int lcol = wn + j * 16 + (lane & 15);
        if (m0 + lrow < count) {
          const float g = accG[i][j][r], u = accU[i][j][r];
          const float h = (g / (1.f + __expf(-g))) * u;   // silu(g) * u
          hidden[(size_t)(hbase + m0 + lrow) * MOE_F + n0 + lcol] = (bf16)h;
        }
      }
}

// ---------------------------------------------------------------------------
// 4) Down-proj grouped GEMM: contrib[token][slot] = gate * (hidden_e @ Wd^T).
//    Same tiling; K = F = 4096 (128 K-steps). Async A staging (hidden is bf16).
// ---------------------------------------------------------------------------
__global__ __launch_bounds__(256)
void moe_down_kernel(const bf16* __restrict__ hidden,
                     const float* __restrict__ Wd, const float* __restrict__ sWd,
                     const int* __restrict__ token_ids, const float* __restrict__ gts,
                     const int* __restrict__ slots,
                     const int* __restrict__ counts, const int* __restrict__ offs,
                     float* __restrict__ contrib) {
  const int e     = blockIdx.z;
  const int count = counts[e];
  const int m0    = blockIdx.y * 128;
  if (m0 >= count) return;
  const int n0 = blockIdx.x * 64;   // column in D

  const float* wd = (e < MOE_E) ? (Wd + (size_t)e * MOE_D * MOE_F) : sWd;
  const int*  ids = token_ids + e * MOE_T;
  const int hbase = offs[e];

  __shared__ __align__(16) bf16 As[2][128 * MOE_LDA];
  __shared__ __align__(16) bf16 Bs[2][ 64 * MOE_LDA];

  const int tid = threadIdx.x;
  auto stage = [&](int buf, int kk) {
    {
      const int row = tid >> 1, seg = (tid & 1) * 16;
      const unsigned loff = moe_lds_off(&As[buf][row * MOE_LDA + seg]);
      if (m0 + row < count) {
        const unsigned voff =
            (unsigned)(((size_t)(hbase + m0 + row) * MOE_F + kk + seg) *
                       sizeof(bf16));
        moe_async_ld128(loff, voff, hidden);
        moe_async_ld128(loff + 16u, voff + 16u, hidden);
      } else {
        const uint4 z = make_uint4(0, 0, 0, 0);
        uint4* q = (uint4*)(&As[buf][row * MOE_LDA + seg]);
        q[0] = z; q[1] = z;
      }
    }
    {
      const int r = tid >> 2, c8 = (tid & 3) * 8;
      const float* pb = wd + (size_t)(n0 + r) * MOE_F + kk + c8;
      v8bf vb;
#pragma unroll
      for (int i = 0; i < 8; ++i) vb[i] = (bf16)pb[i];
      *(v8bf*)(&Bs[buf][r * MOE_LDA + c8]) = vb;
    }
  };

  const int lane = tid & 31;
  const int w  = tid >> 5;
  const int wm = (w >> 1) * 32;
  const int wn = (w & 1) * 32;

  const v8f vzero = {0.f, 0.f, 0.f, 0.f, 0.f, 0.f, 0.f, 0.f};
  v8f acc[2][2];
#pragma unroll
  for (int i = 0; i < 2; ++i)
#pragma unroll
    for (int j = 0; j < 2; ++j) acc[i][j] = vzero;

  auto loadA = [&](int buf, int mt) -> v16bf {
    const int row = wm + mt * 16 + (lane & 15);
    const int kb  = (lane >> 4) * 8;
    const bf16* base = &As[buf][row * MOE_LDA + kb];
    return moe_cat16(*(const v8bf*)(base), *(const v8bf*)(base + 16));
  };
  auto loadB = [&](int buf, int nt) -> v16bf {
    const int rr = wn + nt * 16 + (lane & 15);
    const int kb = (lane >> 4) * 8;
    const bf16* base = &Bs[buf][rr * MOE_LDA + kb];
    return moe_cat16(*(const v8bf*)(base), *(const v8bf*)(base + 16));
  };

  stage(0, 0);
  moe_wait_async();
  __syncthreads();
  const int KS = MOE_F / MOE_KC;   // 128 K-steps
  for (int ks = 0; ks < KS; ++ks) {
    const int cur = ks & 1;
    if (ks + 1 < KS) stage(cur ^ 1, (ks + 1) * MOE_KC);
    v16bf a0 = loadA(cur, 0), a1 = loadA(cur, 1);
    v16bf b0 = loadB(cur, 0), b1 = loadB(cur, 1);
    acc[0][0] = moe_wmma_bf16(a0, b0, acc[0][0]);
    acc[0][1] = moe_wmma_bf16(a0, b1, acc[0][1]);
    acc[1][0] = moe_wmma_bf16(a1, b0, acc[1][0]);
    acc[1][1] = moe_wmma_bf16(a1, b1, acc[1][1]);
    moe_wait_async();
    __syncthreads();
  }

#pragma unroll
  for (int i = 0; i < 2; ++i)
#pragma unroll
    for (int j = 0; j < 2; ++j)
#pragma unroll
      for (int r = 0; r < 8; ++r) {
        const int lrow = wm + i * 16 + r + ((lane >> 4) * 8);
        const int lcol = wn + j * 16 + (lane & 15);
        const int gm = m0 + lrow;
        if (gm < count) {
          const int tok    = ids[gm];
          const float gate = gts[e * MOE_T + gm];
          const int slot   = slots[e * MOE_T + gm];
          contrib[((size_t)tok * 3 + slot) * MOE_D + n0 + lcol] =
              gate * acc[i][j][r];
        }
      }
}

// ---------------------------------------------------------------------------
// 5) Fixed-order combine: out = slot0 + slot1 + shared  (bit-deterministic)
// ---------------------------------------------------------------------------
__global__ __launch_bounds__(256)
void moe_combine_kernel(const float* __restrict__ contrib, float* __restrict__ out) {
  const int gid = blockIdx.x * 256 + threadIdx.x;       // over T*D/4 float4s
  const int t   = gid / (MOE_D / 4);
  const int d4  = gid % (MOE_D / 4);
  const float4* c = (const float4*)contrib;
  const float4 a = c[((size_t)t * 3 + 0) * (MOE_D / 4) + d4];
  const float4 b = c[((size_t)t * 3 + 1) * (MOE_D / 4) + d4];
  const float4 s = c[((size_t)t * 3 + 2) * (MOE_D / 4) + d4];
  float4 o;
  o.x = a.x + b.x + s.x; o.y = a.y + b.y + s.y;
  o.z = a.z + b.z + s.z; o.w = a.w + b.w + s.w;
  ((float4*)out)[(size_t)t * (MOE_D / 4) + d4] = o;
}

// ---------------------------------------------------------------------------
extern "C" void kernel_launch(void* const* d_in, const int* in_sizes, int n_in,
                              void* d_out, int out_size, void* d_ws, size_t ws_size,
                              hipStream_t stream) {
  (void)in_sizes; (void)n_in; (void)out_size; (void)ws_size;
  const float* x   = (const float*)d_in[0];
  const float* Wr  = (const float*)d_in[1];
  const float* Wg  = (const float*)d_in[2];
  const float* Wu  = (const float*)d_in[3];
  const float* Wd  = (const float*)d_in[4];
  const float* sWg = (const float*)d_in[5];
  const float* sWu = (const float*)d_in[6];
  const float* sWd = (const float*)d_in[7];
  float* out = (float*)d_out;

  char* ws = (char*)d_ws;
  size_t off = 0;
  auto alloc = [&](size_t bytes) -> char* {
    char* p = ws + off;
    off += (bytes + 255) & ~(size_t)255;
    return p;
  };
  bf16*  xb      = (bf16*)alloc((size_t)MOE_T * MOE_D * sizeof(bf16));
  int*   tki     = (int*)alloc((size_t)MOE_T * 2 * sizeof(int));
  float* tkg     = (float*)alloc((size_t)MOE_T * 2 * sizeof(float));
  int*   ids     = (int*)alloc((size_t)MOE_NE * MOE_T * sizeof(int));
  float* gts     = (float*)alloc((size_t)MOE_NE * MOE_T * sizeof(float));
  int*   slots   = (int*)alloc((size_t)MOE_NE * MOE_T * sizeof(int));
  int*   counts  = (int*)alloc(16 * sizeof(int));
  int*   offs    = (int*)alloc(16 * sizeof(int));
  bf16*  hidden  = (bf16*)alloc((size_t)3 * MOE_T * MOE_F * sizeof(bf16));  // 2T routed + T shared rows
  float* contrib = (float*)alloc((size_t)MOE_T * 3 * MOE_D * sizeof(float));

  moe_router_kernel<<<MOE_T / 8, 256, 0, stream>>>(x, Wr, xb, tki, tkg);
  moe_scatter_kernel<<<MOE_NE, 32, 0, stream>>>(tki, tkg, ids, gts, slots, counts);
  moe_offsets_kernel<<<1, 32, 0, stream>>>(counts, offs);

  dim3 g1(MOE_F / 64, MOE_T / 128, MOE_NE);
  moe_gateup_kernel<<<g1, 256, 0, stream>>>(xb, Wg, Wu, sWg, sWu, ids, counts, offs,
                                            hidden);
  dim3 g2(MOE_D / 64, MOE_T / 128, MOE_NE);
  moe_down_kernel<<<g2, 256, 0, stream>>>(hidden, Wd, sWd, ids, gts, slots, counts,
                                          offs, contrib);
  moe_combine_kernel<<<(MOE_T * MOE_D / 4) / 256, 256, 0, stream>>>(contrib, out);
}
